// GreedyDecoder_6270652253006
// MI455X (gfx1250) — compile-verified, matrix operand order checked by reading, not run on hardware
//
#include <hip/hip_runtime.h>
#include <hip/hip_bf16.h>
#include <math.h>

#define L_DEC   24
#define BATCH   4
#define DM      1536
#define NH      8
#define DHD     192
#define FFD     2048
#define VOC     50257
#define EOS_TOK 50256
#define TENC    48
#define DENC    512
#define WEMB    768
#define NEGB    (-1e9f)
#define NLAY    2

typedef __attribute__((ext_vector_type(16))) _Float16 v16h;
typedef __attribute__((ext_vector_type(8)))  _Float16 v8h;
typedef __attribute__((ext_vector_type(4)))  _Float16 v4h;
typedef __attribute__((ext_vector_type(8)))  float    v8f;

// ---------------------------------------------------------------------------
// Fragment loaders (CDNA5 ISA 7.12.2 layout):
//   16-bit A 16x32: lane l holds row m0+(l&15); halves {kb..kb+7},{kb+16..kb+23},
//   kb = k + 8*(l>=16). B mirrored with N in place of M (W stored [N,K]).
// ---------------------------------------------------------------------------
__device__ __forceinline__ void load_frag_f32cvt(const float* __restrict__ p, int kb, v16h& f) {
  const float4* q0 = (const float4*)(p + kb);
  const float4* q1 = (const float4*)(p + kb + 16);
  float4 x0 = q0[0], x1 = q0[1], x2 = q1[0], x3 = q1[1];
  f[0]=(_Float16)x0.x; f[1]=(_Float16)x0.y; f[2]=(_Float16)x0.z; f[3]=(_Float16)x0.w;
  f[4]=(_Float16)x1.x; f[5]=(_Float16)x1.y; f[6]=(_Float16)x1.z; f[7]=(_Float16)x1.w;
  f[8]=(_Float16)x2.x; f[9]=(_Float16)x2.y; f[10]=(_Float16)x2.z; f[11]=(_Float16)x2.w;
  f[12]=(_Float16)x3.x; f[13]=(_Float16)x3.y; f[14]=(_Float16)x3.z; f[15]=(_Float16)x3.w;
}

__device__ __forceinline__ v16h load_frag_h(const _Float16* __restrict__ p, int kb) {
  v8h lo = *(const v8h*)(p + kb);        // 16B aligned (kb multiple of 8, rows mult of 8)
  v8h hi = *(const v8h*)(p + kb + 16);
  v16h f;
  #pragma unroll
  for (int j = 0; j < 8; ++j) { f[j] = lo[j]; f[j + 8] = hi[j]; }
  return f;
}

// ---------------------------------------------------------------------------
// fp32-weight WMMA GEMM (fallback + one-time preamble GEMMs)
// C[M,N] = A[M,K] * W[N,K]^T (+bias)(opt relu); one wave per 16x16 tile.
// ---------------------------------------------------------------------------
__global__ __launch_bounds__(32) void k_gemm(
    const float* __restrict__ A, int lda,
    const float* __restrict__ W, int ldw,
    const float* __restrict__ bias,
    float* __restrict__ C, int ldc,
    int M, int N, int K, int relu)
{
  const int lane = threadIdx.x;
  const int hs = (lane >> 4) & 1;
  const int rc = lane & 15;
  const int m0 = blockIdx.x << 4;
  const int n0 = blockIdx.y << 4;
  int ar = m0 + rc; if (ar > M - 1) ar = M - 1;
  int wr = n0 + rc; if (wr > N - 1) wr = N - 1;
  const float* Ap = A + (size_t)ar * lda;
  const float* Wp = W + (size_t)wr * ldw;
  v8f acc = {};
  #pragma unroll 2
  for (int k = 0; k < K; k += 32) {
    const int kb = k + (hs << 3);
    v16h a, b;
    load_frag_f32cvt(Ap, kb, a);
    load_frag_f32cvt(Wp, kb, b);
    acc = __builtin_amdgcn_wmma_f32_16x16x32_f16(false, a, false, b,
                                                 (short)0, acc, false, false);
  }
  const int col = n0 + rc;
  if (col < N) {
    const float bv = bias ? bias[col] : 0.0f;
    #pragma unroll
    for (int v = 0; v < 8; ++v) {
      const int row = m0 + v + (hs << 3);
      if (row < M) {
        float val = acc[v] + bv;
        if (relu && val < 0.0f) val = 0.0f;
        C[(size_t)row * ldc + col] = val;
      }
    }
  }
}

// ---------------------------------------------------------------------------
// f16-weight WMMA GEMM: 16x64 C strip per wave; A fragment (fp32->f16 inline,
// A is tiny and L2-hot) reused across 4 B fragments -> 4 WMMA per K step.
// Weight stream prefetched (global_prefetch_b8).
// ---------------------------------------------------------------------------
__global__ __launch_bounds__(32) void k_gemm_h(
    const float* __restrict__ A, int lda,
    const _Float16* __restrict__ Wh, int ldw,
    const float* __restrict__ bias,
    float* __restrict__ C, int ldc,
    int M, int N, int K, int relu)
{
  const int lane = threadIdx.x;
  const int hs = (lane >> 4) & 1;
  const int rc = lane & 15;
  const int m0 = blockIdx.x << 4;
  const int n0 = blockIdx.y << 6;
  int ar = m0 + rc; if (ar > M - 1) ar = M - 1;
  const float* Ap = A + (size_t)ar * lda;
  const _Float16* Wp0; const _Float16* Wp1; const _Float16* Wp2; const _Float16* Wp3;
  {
    int w0 = n0 + rc;      if (w0 > N - 1) w0 = N - 1;
    int w1 = n0 + 16 + rc; if (w1 > N - 1) w1 = N - 1;
    int w2 = n0 + 32 + rc; if (w2 > N - 1) w2 = N - 1;
    int w3 = n0 + 48 + rc; if (w3 > N - 1) w3 = N - 1;
    Wp0 = Wh + (size_t)w0 * ldw; Wp1 = Wh + (size_t)w1 * ldw;
    Wp2 = Wh + (size_t)w2 * ldw; Wp3 = Wh + (size_t)w3 * ldw;
  }
  v8f acc0 = {}, acc1 = {}, acc2 = {}, acc3 = {};
  for (int k = 0; k < K; k += 32) {
    const int kb = k + (hs << 3);
    __builtin_prefetch((const void*)(Wp0 + kb + 256), 0, 0);
    __builtin_prefetch((const void*)(Wp1 + kb + 256), 0, 0);
    __builtin_prefetch((const void*)(Wp2 + kb + 256), 0, 0);
    __builtin_prefetch((const void*)(Wp3 + kb + 256), 0, 0);
    v16h a;
    load_frag_f32cvt(Ap, kb, a);
    v16h b0 = load_frag_h(Wp0, kb);
    v16h b1 = load_frag_h(Wp1, kb);
    v16h b2 = load_frag_h(Wp2, kb);
    v16h b3 = load_frag_h(Wp3, kb);
    acc0 = __builtin_amdgcn_wmma_f32_16x16x32_f16(false, a, false, b0, (short)0, acc0, false, false);
    acc1 = __builtin_amdgcn_wmma_f32_16x16x32_f16(false, a, false, b1, (short)0, acc1, false, false);
    acc2 = __builtin_amdgcn_wmma_f32_16x16x32_f16(false, a, false, b2, (short)0, acc2, false, false);
    acc3 = __builtin_amdgcn_wmma_f32_16x16x32_f16(false, a, false, b3, (short)0, acc3, false, false);
  }
  #pragma unroll
  for (int j = 0; j < 4; ++j) {
    const v8f& acc = (j == 0) ? acc0 : (j == 1) ? acc1 : (j == 2) ? acc2 : acc3;
    const int col = n0 + j * 16 + rc;
    if (col < N) {
      const float bv = bias ? bias[col] : 0.0f;
      #pragma unroll
      for (int v = 0; v < 8; ++v) {
        const int row = m0 + v + (hs << 3);
        if (row < M) {
          float val = acc[v] + bv;
          if (relu && val < 0.0f) val = 0.0f;
          C[(size_t)row * ldc + col] = val;
        }
      }
    }
  }
}

// ---------------------------------------------------------------------------
// fp32 -> f16 weight conversion (vectorized, n multiple of 4)
// ---------------------------------------------------------------------------
__global__ void k_f32_to_f16(const float* __restrict__ src, _Float16* __restrict__ dst, int n4) {
  int i = blockIdx.x * 256 + threadIdx.x;
  if (i < n4) {
    float4 v = ((const float4*)src)[i];
    v4h o; o[0] = (_Float16)v.x; o[1] = (_Float16)v.y; o[2] = (_Float16)v.z; o[3] = (_Float16)v.w;
    ((v4h*)dst)[i] = o;
  }
}

// ---------------------------------------------------------------------------
// Init kernels
// ---------------------------------------------------------------------------
__global__ void k_pe(float* __restrict__ pe) {
  int i = blockIdx.x * 256 + threadIdx.x;            // L_DEC*DM
  if (i < L_DEC * DM) {
    int p = i / DM, d = i % DM;
    int pair = d >> 1;
    float div = __expf((float)(2 * pair) * (-logf(10000.0f) / (float)DM));
    float ang = (float)p * div;
    pe[i] = (d & 1) ? cosf(ang) : sinf(ang);
  }
}

__global__ void k_init_built(const float* __restrict__ embed,
                             float* __restrict__ built, int* __restrict__ dmask) {
  int i = blockIdx.x * 256 + threadIdx.x;            // L_DEC*BATCH*DM
  if (i < L_DEC * BATCH * DM) {
    int s = i / (BATCH * DM);
    int d = i % DM;
    float v = 0.0f;
    if (s == 0 && d < WEMB) v = embed[(size_t)EOS_TOK * WEMB + d];
    built[i] = v;
  }
  if (i < L_DEC * BATCH) dmask[i] = (i < BATCH) ? 1 : 0;   // dmask[0,b]=True
}

__global__ void k_logits0(float* __restrict__ out) {
  int i = blockIdx.x * 256 + threadIdx.x;            // BATCH*VOC
  if (i < BATCH * VOC) {
    int v = i % VOC;
    out[i] = (v == EOS_TOK) ? 1.0f : 0.0f;
  }
}

__global__ void k_add_pe(const float* __restrict__ built, const float* __restrict__ pe,
                         float* __restrict__ x) {
  int i = blockIdx.x * 256 + threadIdx.x;            // L_DEC*BATCH*DM
  if (i < L_DEC * BATCH * DM) {
    int s = i / (BATCH * DM);
    int d = i % DM;
    x[i] = built[i] + pe[s * DM + d];
  }
}

// ---------------------------------------------------------------------------
// LayerNorm(residual + h) per row of DM. In-place safe (values cached in regs).
// ---------------------------------------------------------------------------
__global__ __launch_bounds__(256) void k_ln(const float* __restrict__ res,
                                            const float* __restrict__ h,
                                            const float* __restrict__ g,
                                            const float* __restrict__ b,
                                            float* __restrict__ out) {
  const int row = blockIdx.x;
  const int t = threadIdx.x;
  const float* r = res + (size_t)row * DM;
  const float* hh = h + (size_t)row * DM;
  float vals[DM / 256];
  float s = 0.0f, s2 = 0.0f;
  #pragma unroll
  for (int j = 0; j < DM / 256; ++j) {
    float v = r[t + j * 256] + hh[t + j * 256];
    vals[j] = v; s += v; s2 += v * v;
  }
  #pragma unroll
  for (int off = 16; off > 0; off >>= 1) {
    s  += __shfl_down(s, off, 32);
    s2 += __shfl_down(s2, off, 32);
  }
  __shared__ float sh[8], sh2[8];
  const int wid = t >> 5;
  if ((t & 31) == 0) { sh[wid] = s; sh2[wid] = s2; }
  __syncthreads();
  if (t == 0) {
    float S = 0.0f, S2 = 0.0f;
    for (int w = 0; w < 8; ++w) { S += sh[w]; S2 += sh2[w]; }
    sh[0] = S; sh2[0] = S2;
  }
  __syncthreads();
  const float mu = sh[0] / (float)DM;
  const float var = sh2[0] / (float)DM - mu * mu;
  const float inv = rsqrtf(var + 1e-5f);
  #pragma unroll
  for (int j = 0; j < DM / 256; ++j) {
    int c = t + j * 256;
    out[(size_t)row * DM + c] = (vals[j] - mu) * inv * g[c] + b[c];
  }
}

// ---------------------------------------------------------------------------
// Self-attention (qkv rows = s*BATCH+b, ld=3*DM; q@0,k@DM,v@2*DM); block=(b,h,q)
// ---------------------------------------------------------------------------
__global__ __launch_bounds__(64) void k_attn_self(const float* __restrict__ qkv,
                                                  const int* __restrict__ dmask,
                                                  int step, float* __restrict__ out) {
  const int q = blockIdx.x % L_DEC;
  const int hh = (blockIdx.x / L_DEC) % NH;
  const int b = blockIdx.x / (L_DEC * NH);
  const float* qp = qkv + ((size_t)(q * BATCH + b)) * (3 * DM) + hh * DHD;
  __shared__ float sc[L_DEC];
  const int t = threadIdx.x;
  if (t < L_DEC) {
    const int k = t;
    const bool ok = (k <= q) && (k < step) && (dmask[k * BATCH + b] != 0);
    float s = NEGB;
    if (ok) {
      const float* kp = qkv + ((size_t)(k * BATCH + b)) * (3 * DM) + DM + hh * DHD;
      float acc = 0.0f;
      for (int d = 0; d < DHD; ++d) acc = fmaf(qp[d], kp[d], acc);
      s = acc * 0.0721687836f;   // 1/sqrt(192)
    }
    sc[k] = s;
  }
  __syncthreads();
  if (t == 0) {
    float mx = sc[0];
    for (int k = 1; k < L_DEC; ++k) mx = fmaxf(mx, sc[k]);
    float sum = 0.0f;
    for (int k = 0; k < L_DEC; ++k) { float e = __expf(sc[k] - mx); sc[k] = e; sum += e; }
    float r = 1.0f / sum;
    for (int k = 0; k < L_DEC; ++k) sc[k] *= r;
  }
  __syncthreads();
  for (int d = t; d < DHD; d += 64) {
    float acc = 0.0f;
    for (int k = 0; k < L_DEC; ++k) {
      const float* vp = qkv + ((size_t)(k * BATCH + b)) * (3 * DM) + 2 * DM + hh * DHD;
      acc = fmaf(sc[k], vp[d], acc);
    }
    out[((size_t)(q * BATCH + b)) * DM + hh * DHD + d] = acc;
  }
}

// Cross-attention: q in qbuf (ld 3*DM, off 0), kv rows t*BATCH+b ld 2*DM (k@0,v@DM)
__global__ __launch_bounds__(64) void k_attn_cross(const float* __restrict__ qbuf,
                                                   const float* __restrict__ kvmem,
                                                   const unsigned char* __restrict__ pad,
                                                   float* __restrict__ out) {
  const int q = blockIdx.x % L_DEC;
  const int hh = (blockIdx.x / L_DEC) % NH;
  const int b = blockIdx.x / (L_DEC * NH);
  const float* qp = qbuf + ((size_t)(q * BATCH + b)) * (3 * DM) + hh * DHD;
  __shared__ float sc[TENC];
  const int t = threadIdx.x;
  if (t < TENC) {
    const float* kp = kvmem + ((size_t)(t * BATCH + b)) * (2 * DM) + hh * DHD;
    float acc = 0.0f;
    for (int d = 0; d < DHD; ++d) acc = fmaf(qp[d], kp[d], acc);
    float s = acc * 0.0721687836f;
    if (pad[b * TENC + t]) s += NEGB;
    sc[t] = s;
  }
  __syncthreads();
  if (t == 0) {
    float mx = sc[0];
    for (int k = 1; k < TENC; ++k) mx = fmaxf(mx, sc[k]);
    float sum = 0.0f;
    for (int k = 0; k < TENC; ++k) { float e = __expf(sc[k] - mx); sc[k] = e; sum += e; }
    float r = 1.0f / sum;
    for (int k = 0; k < TENC; ++k) sc[k] *= r;
  }
  __syncthreads();
  for (int d = t; d < DHD; d += 64) {
    float acc = 0.0f;
    for (int k = 0; k < TENC; ++k) {
      const float* vp = kvmem + ((size_t)(k * BATCH + b)) * (2 * DM) + DM + hh * DHD;
      acc = fmaf(sc[k], vp[d], acc);
    }
    out[((size_t)(q * BATCH + b)) * DM + hh * DHD + d] = acc;
  }
}

// ---------------------------------------------------------------------------
// argmax per batch row (first-occurrence tie-break like jnp.argmax)
// ---------------------------------------------------------------------------
__global__ __launch_bounds__(256) void k_argmax(const float* __restrict__ logits,
                                                int* __restrict__ token) {
  const int b = blockIdx.x;
  const float* p = logits + (size_t)b * VOC;
  float best = -3.4e38f; int bi = 0x7fffffff;
  for (int v = threadIdx.x; v < VOC; v += 256) {
    float x = p[v];
    if (x > best) { best = x; bi = v; }
  }
  __shared__ float sv[256]; __shared__ int si[256];
  sv[threadIdx.x] = best; si[threadIdx.x] = bi;
  __syncthreads();
  for (int s = 128; s > 0; s >>= 1) {
    if (threadIdx.x < s) {
      float o = sv[threadIdx.x + s]; int oi = si[threadIdx.x + s];
      if (o > sv[threadIdx.x] || (o == sv[threadIdx.x] && oi < si[threadIdx.x])) {
        sv[threadIdx.x] = o; si[threadIdx.x] = oi;
      }
    }
    __syncthreads();
  }
  if (threadIdx.x == 0) token[b] = si[0];
}

__global__ void k_post(const int* __restrict__ token, int* __restrict__ dmask, int step) {
  const int b = threadIdx.x;
  if (b < BATCH) {
    int alive = (token[b] != EOS_TOK) ? 1 : 0;
    dmask[step * BATCH + b] = alive & dmask[(step - 1) * BATCH + b];
  }
}

__global__ void k_gather(const float* __restrict__ embed, const int* __restrict__ token,
                         float* __restrict__ lmA) {
  int i = blockIdx.x * 256 + threadIdx.x;   // BATCH*WEMB
  if (i < BATCH * WEMB) {
    int b = i / WEMB, d = i % WEMB;
    lmA[i] = embed[(size_t)token[b] * WEMB + d];
  }
}

__global__ void k_write_row(const float* __restrict__ decoded, const float* __restrict__ lmfeat,
                            float* __restrict__ built, int step) {
  int i = blockIdx.x * 256 + threadIdx.x;   // BATCH*DM
  if (i < BATCH * DM) {
    int b = i / DM, d = i % DM;
    float v = (d < WEMB) ? decoded[b * WEMB + d] : lmfeat[b * WEMB + (d - WEMB)];
    built[((size_t)step * BATCH + b) * DM + d] = v;
  }
}

__global__ void k_dmask_out(const int* __restrict__ dmask, float* __restrict__ out) {
  int i = threadIdx.x;
  if (i < L_DEC * BATCH) out[(size_t)L_DEC * BATCH * VOC + i] = (float)dmask[i];
}

// ---------------------------------------------------------------------------
// Host launcher
// ---------------------------------------------------------------------------
static inline void gemm32(hipStream_t s, const float* A, int lda, const float* W, int ldw,
                          const float* bias, float* C, int ldc, int M, int N, int K,
                          int relu = 0) {
  dim3 g((M + 15) / 16, (N + 15) / 16);
  k_gemm<<<g, dim3(32), 0, s>>>(A, lda, W, ldw, bias, C, ldc, M, N, K, relu);
}

static inline void gemm16(hipStream_t s, const float* A, int lda, const _Float16* Wh, int ldw,
                          const float* bias, float* C, int ldc, int M, int N, int K,
                          int relu = 0) {
  dim3 g((M + 15) / 16, (N + 63) / 64);
  k_gemm_h<<<g, dim3(32), 0, s>>>(A, lda, Wh, ldw, bias, C, ldc, M, N, K, relu);
}

static inline void cvt16(hipStream_t s, const float* src, _Float16* dst, size_t n) {
  int n4 = (int)(n / 4);
  k_f32_to_f16<<<(n4 + 255) / 256, 256, 0, s>>>(src, dst, n4);
}

extern "C" void kernel_launch(void* const* d_in, const int* in_sizes, int n_in,
                              void* d_out, int out_size, void* d_ws, size_t ws_size,
                              hipStream_t stream) {
  (void)in_sizes; (void)n_in; (void)out_size;
  const float* enc_out  = (const float*)d_in[0];
  const unsigned char* enc_pad = (const unsigned char*)d_in[1];
  const float* embed    = (const float*)d_in[2];
  const float* W_lm     = (const float*)d_in[3];
  const float* b_lm     = (const float*)d_in[4];
  const float* enc_w    = (const float*)d_in[5];
  const float* enc_b    = (const float*)d_in[6];
  const float* bridge_w = (const float*)d_in[7];
  const float* bridge_b = (const float*)d_in[8];
  const float* sa_in_w  = (const float*)d_in[9];
  const float* sa_in_b  = (const float*)d_in[10];
  const float* sa_out_w = (const float*)d_in[11];
  const float* sa_out_b = (const float*)d_in[12];
  const float* ca_in_w  = (const float*)d_in[13];
  const float* ca_in_b  = (const float*)d_in[14];
  const float* ca_out_w = (const float*)d_in[15];
  const float* ca_out_b = (const float*)d_in[16];
  const float* ff1_w    = (const float*)d_in[17];
  const float* ff1_b    = (const float*)d_in[18];
  const float* ff2_w    = (const float*)d_in[19];
  const float* ff2_b    = (const float*)d_in[20];
  const float* ln1_g    = (const float*)d_in[21];
  const float* ln1_b    = (const float*)d_in[22];
  const float* ln2_g    = (const float*)d_in[23];
  const float* ln2_b    = (const float*)d_in[24];
  const float* ln3_g    = (const float*)d_in[25];
  const float* ln3_b    = (const float*)d_in[26];

  float* out = (float*)d_out;

  // ---- fp32 workspace carve-up -----------------------------------------
  float* ws = (float*)d_ws;
  float* pe      = ws;  ws += L_DEC * DM;
  float* memory  = ws;  ws += TENC * BATCH * DM;
  float* kvmem   = ws;  ws += NLAY * TENC * BATCH * 2 * DM;
  float* built   = ws;  ws += L_DEC * BATCH * DM;
  float* xbuf    = ws;  ws += L_DEC * BATCH * DM;
  float* x2      = ws;  ws += L_DEC * BATCH * DM;
  float* qkv     = ws;  ws += L_DEC * BATCH * 3 * DM;
  float* attnout = ws;  ws += L_DEC * BATCH * DM;
  float* h1      = ws;  ws += L_DEC * BATCH * FFD;
  float* h2      = ws;  ws += L_DEC * BATCH * DM;
  float* decoded = ws;  ws += BATCH * WEMB;
  float* lmA     = ws;  ws += BATCH * WEMB;
  float* lmfeat  = ws;  ws += BATCH * WEMB;
  int*   dmask   = (int*)ws;  ws += L_DEC * BATCH;
  int*   token   = (int*)ws;  ws += BATCH;

  // ---- f16 weight cache carve-up (used only if workspace is big enough) --
  _Float16* hp = (_Float16*)ws;
  _Float16* sa_in_h  = hp;  hp += (size_t)NLAY * 3 * DM * DM;
  _Float16* sa_out_h = hp;  hp += (size_t)NLAY * DM * DM;
  _Float16* ca_inq_h = hp;  hp += (size_t)NLAY * DM * DM;     // q rows only
  _Float16* ca_out_h = hp;  hp += (size_t)NLAY * DM * DM;
  _Float16* ff1_h    = hp;  hp += (size_t)NLAY * FFD * DM;
  _Float16* ff2_h    = hp;  hp += (size_t)NLAY * DM * FFD;
  _Float16* bridge_h = hp;  hp += (size_t)WEMB * DM;
  _Float16* wlm_h    = hp;  hp += (size_t)WEMB * WEMB;
  _Float16* embed_h  = hp;  hp += (size_t)VOC * WEMB;
  const size_t need = (size_t)((char*)hp - (char*)d_ws);
  const bool useH = (ws_size >= need);   // ws_size fixed per run -> deterministic

  const int ROWS = L_DEC * BATCH;            // 96
  const int XN = L_DEC * BATCH * DM;         // 147456

  // ---- one-time preamble -------------------------------------------------
  k_pe<<<(L_DEC * DM + 255) / 256, 256, 0, stream>>>(pe);
  k_init_built<<<(XN + 255) / 256, 256, 0, stream>>>(embed, built, dmask);
  k_logits0<<<(BATCH * VOC + 255) / 256, 256, 0, stream>>>(out);

  if (useH) {
    cvt16(stream, sa_in_w,  sa_in_h,  (size_t)NLAY * 3 * DM * DM);
    cvt16(stream, sa_out_w, sa_out_h, (size_t)NLAY * DM * DM);
    for (int l = 0; l < NLAY; ++l)   // q rows of ca_in_w are layer-contiguous
      cvt16(stream, ca_in_w + (size_t)l * 3 * DM * DM, ca_inq_h + (size_t)l * DM * DM,
            (size_t)DM * DM);
    cvt16(stream, ca_out_w, ca_out_h, (size_t)NLAY * DM * DM);
    cvt16(stream, ff1_w,    ff1_h,    (size_t)NLAY * FFD * DM);
    cvt16(stream, ff2_w,    ff2_h,    (size_t)NLAY * DM * FFD);
    cvt16(stream, bridge_w, bridge_h, (size_t)WEMB * DM);
    cvt16(stream, W_lm,     wlm_h,    (size_t)WEMB * WEMB);
    cvt16(stream, embed,    embed_h,  (size_t)VOC * WEMB);
  }

  // memory = encoder_output @ enc_w^T + enc_b   [T*B, DM]  (one-time, fp32 path)
  gemm32(stream, enc_out, DENC, enc_w, DENC, enc_b, memory, DM,
         TENC * BATCH, DM, DENC);
  // cross-attn K,V once per layer
  for (int l = 0; l < NLAY; ++l) {
    gemm32(stream, memory, DM,
           ca_in_w + (size_t)l * 3 * DM * DM + (size_t)DM * DM, DM,
           ca_in_b + (size_t)l * 3 * DM + DM,
           kvmem + (size_t)l * TENC * BATCH * 2 * DM, 2 * DM,
           TENC * BATCH, 2 * DM, DM);
  }

  // ---- sequential greedy decode -----------------------------------------
  for (int i = 1; i < L_DEC; ++i) {
    k_add_pe<<<(XN + 255) / 256, 256, 0, stream>>>(built, pe, xbuf);

    for (int l = 0; l < NLAY; ++l) {
      // self-attention
      if (useH)
        gemm16(stream, xbuf, DM, sa_in_h + (size_t)l * 3 * DM * DM, DM,
               sa_in_b + (size_t)l * 3 * DM, qkv, 3 * DM, ROWS, 3 * DM, DM);
      else
        gemm32(stream, xbuf, DM, sa_in_w + (size_t)l * 3 * DM * DM, DM,
               sa_in_b + (size_t)l * 3 * DM, qkv, 3 * DM, ROWS, 3 * DM, DM);
      k_attn_self<<<BATCH * NH * L_DEC, 64, 0, stream>>>(qkv, dmask, i, attnout);
      if (useH)
        gemm16(stream, attnout, DM, sa_out_h + (size_t)l * DM * DM, DM,
               sa_out_b + (size_t)l * DM, h2, DM, ROWS, DM, DM);
      else
        gemm32(stream, attnout, DM, sa_out_w + (size_t)l * DM * DM, DM,
               sa_out_b + (size_t)l * DM, h2, DM, ROWS, DM, DM);
      k_ln<<<ROWS, 256, 0, stream>>>(xbuf, h2, ln1_g + l * DM, ln1_b + l * DM, x2);

      // cross-attention (q projection only; K,V precomputed)
      if (useH)
        gemm16(stream, x2, DM, ca_inq_h + (size_t)l * DM * DM, DM,
               ca_in_b + (size_t)l * 3 * DM, qkv, 3 * DM, ROWS, DM, DM);
      else
        gemm32(stream, x2, DM, ca_in_w + (size_t)l * 3 * DM * DM, DM,
               ca_in_b + (size_t)l * 3 * DM, qkv, 3 * DM, ROWS, DM, DM);
      k_attn_cross<<<BATCH * NH * L_DEC, 64, 0, stream>>>(
          qkv, kvmem + (size_t)l * TENC * BATCH * 2 * DM, enc_pad, attnout);
      if (useH)
        gemm16(stream, attnout, DM, ca_out_h + (size_t)l * DM * DM, DM,
               ca_out_b + (size_t)l * DM, h2, DM, ROWS, DM, DM);
      else
        gemm32(stream, attnout, DM, ca_out_w + (size_t)l * DM * DM, DM,
               ca_out_b + (size_t)l * DM, h2, DM, ROWS, DM, DM);
      k_ln<<<ROWS, 256, 0, stream>>>(x2, h2, ln2_g + l * DM, ln2_b + l * DM, xbuf);

      // FFN
      if (useH) {
        gemm16(stream, xbuf, DM, ff1_h + (size_t)l * FFD * DM, DM,
               ff1_b + (size_t)l * FFD, h1, FFD, ROWS, FFD, DM, /*relu=*/1);
        gemm16(stream, h1, FFD, ff2_h + (size_t)l * DM * FFD, FFD,
               ff2_b + (size_t)l * DM, h2, DM, ROWS, DM, FFD);
      } else {
        gemm32(stream, xbuf, DM, ff1_w + (size_t)l * FFD * DM, DM,
               ff1_b + (size_t)l * FFD, h1, FFD, ROWS, FFD, DM, /*relu=*/1);
        gemm32(stream, h1, FFD, ff2_w + (size_t)l * DM * FFD, FFD,
               ff2_b + (size_t)l * DM, h2, DM, ROWS, DM, FFD);
      }
      k_ln<<<ROWS, 256, 0, stream>>>(xbuf, h2, ln3_g + l * DM, ln3_b + l * DM, xbuf);
    }

    // bridge on row i-1: [4, WEMB]
    if (useH)
      gemm16(stream, xbuf + (size_t)(i - 1) * BATCH * DM, DM, bridge_h, DM,
             bridge_b, decoded, WEMB, BATCH, WEMB, DM);
    else
      gemm32(stream, xbuf + (size_t)(i - 1) * BATCH * DM, DM, bridge_w, DM,
             bridge_b, decoded, WEMB, BATCH, WEMB, DM);
    // logits = decoded @ embed^T  -> out[i]   (dominant byte stream: f16 embed)
    if (useH)
      gemm16(stream, decoded, WEMB, embed_h, WEMB, nullptr,
             out + (size_t)i * BATCH * VOC, VOC, BATCH, VOC, WEMB);
    else
      gemm32(stream, decoded, WEMB, embed, WEMB, nullptr,
             out + (size_t)i * BATCH * VOC, VOC, BATCH, VOC, WEMB);
    k_argmax<<<BATCH, 256, 0, stream>>>(out + (size_t)i * BATCH * VOC, token);
    k_post<<<1, 32, 0, stream>>>(token, dmask, i);
    // lm_feat = embed[token] @ W_lm^T + b_lm
    k_gather<<<(BATCH * WEMB + 255) / 256, 256, 0, stream>>>(embed, token, lmA);
    if (useH)
      gemm16(stream, lmA, WEMB, wlm_h, WEMB, b_lm, lmfeat, WEMB, BATCH, WEMB, WEMB);
    else
      gemm32(stream, lmA, WEMB, W_lm, WEMB, b_lm, lmfeat, WEMB, BATCH, WEMB, WEMB);
    k_write_row<<<(BATCH * DM + 255) / 256, 256, 0, stream>>>(decoded, lmfeat, built, i);
  }

  k_dmask_out<<<1, 128, 0, stream>>>(dmask, out);
}
